// QGaloreLinear_33689723470518
// MI455X (gfx1250) — matrix operand order repaired and checked by reading, not compiled, not run on hardware
//
#include <hip/hip_runtime.h>
#include <hip/hip_bf16.h>

// ---------------------------------------------------------------------------
// Dequantized-weight GEMM for MI455X (gfx1250, wave32, WMMA).
//   y[m,n] = sum_k x[m,k] * (qw[n,k] - zero[g]) * scale[g] + bias[n]
//   M = 8192, N = 4096, K = 4096, quant group = 256 along flat weight.
//
// Roofline: 275 GFLOP vs ~336 MB HBM traffic (~14us @ 23.3 TB/s) -> compute-
// bound on the matrix pipe. bf16 WMMA (16x16x32) with f32 accum is the widest
// sensible op. Double-buffered LDS + register staging: one barrier per 64-K
// stage, global loads for stage s+1 overlap the 16 WMMAs of stage s, dequant
// VALU co-executes with the WMMA (TRANS) pipe.
// ---------------------------------------------------------------------------

typedef __attribute__((ext_vector_type(4)))  float  v4f;
typedef __attribute__((ext_vector_type(4)))  int    v4i;
typedef __attribute__((ext_vector_type(4)))  __bf16 v4bf;
typedef __attribute__((ext_vector_type(8)))  __bf16 v8bf;
typedef __attribute__((ext_vector_type(16))) __bf16 v16bf;
typedef __attribute__((ext_vector_type(8)))  float  v8f;

constexpr int M_DIM = 4 * 2048;    // B*S
constexpr int N_DIM = 4096;        // OUT
constexpr int K_DIM = 4096;        // IN
constexpr int GROUPS_PER_ROW = K_DIM >> 8;   // 16 groups of 256 per weight row

constexpr int BM = 128;            // block tile M
constexpr int BN = 128;            // block tile N
constexpr int BK = 64;             // k-stage (two bf16 WMMA K-steps)
constexpr int LDS_PITCH = BK + 8;  // 72 halfs = 144B pitch -> conflict-free b128
constexpr int THREADS = 256;       // 8 waves (wave32)
constexpr int CHUNKS = (BM * BK / 4) / THREADS;  // 8 4-elt chunks per thread

__global__ __launch_bounds__(THREADS)
void qlinear_wmma_bf16_kernel(const float* __restrict__ x,
                              const int*   __restrict__ qw,
                              const float* __restrict__ scales,
                              const float* __restrict__ zeros,
                              const float* __restrict__ bias,
                              float*       __restrict__ out)
{
    __shared__ __bf16 As[2][BM * LDS_PITCH];  // x tiles, bf16
    __shared__ __bf16 Ws[2][BN * LDS_PITCH];  // dequantized weight tiles, bf16

    const int tid  = threadIdx.x;
    const int lane = tid & 31;
    const int wave = tid >> 5;
    const int half = lane >> 4;    // 0: lanes 0-15, 1: lanes 16-31
    const int l16  = lane & 15;

    const int m0 = blockIdx.y * BM;
    const int n0 = blockIdx.x * BN;

    const int warp_m = (wave & 1) * 64;   // 2 waves tile M (64 rows each)
    const int warp_n = (wave >> 1) * 32;  // 4 waves tile N (32 cols each)

    v8f acc[4][2];
#pragma unroll
    for (int mi = 0; mi < 4; ++mi)
#pragma unroll
        for (int ni = 0; ni < 2; ++ni)
            acc[mi][ni] = (v8f)(0.0f);

    // Register staging for the next k-stage (raw; converted at LDS-store time
    // so the global-load latency overlaps the WMMA stream).
    v4f   xr[CHUNKS];
    v4i   qr[CHUNKS];
    float sc[CHUNKS], zp[CHUNKS];

    // Chunk mapping: 2048 4-elt chunks per 128x64 tile, 8 per thread.
    // chunk c: row = c >> 4 (16 chunks per 64-wide row), col = (c & 15) * 4.
    auto stage_load = [&](int kt) {
#pragma unroll
        for (int i = 0; i < CHUNKS; ++i) {
            const int c   = tid + i * THREADS;
            const int row = c >> 4;
            const int col = (c & 15) << 2;
            xr[i] = *(const v4f*)(x + (size_t)(m0 + row) * K_DIM + kt + col);
            const int nrow = n0 + row;
            qr[i] = *(const v4i*)(qw + (size_t)nrow * K_DIM + kt + col);
            // BK=64 stage never crosses a 256-elt group: g depends on row only.
            const int g = nrow * GROUPS_PER_ROW + (kt >> 8);
            sc[i] = scales[g];
            zp[i] = zeros[g];
        }
    };

    auto stage_store = [&](int buf) {
#pragma unroll
        for (int i = 0; i < CHUNKS; ++i) {
            const int c   = tid + i * THREADS;
            const int row = c >> 4;
            const int col = (c & 15) << 2;
            *(v4bf*)(&As[buf][row * LDS_PITCH + col]) =
                __builtin_convertvector(xr[i], v4bf);
            v4f wf;
            wf[0] = ((float)qr[i][0] - zp[i]) * sc[i];
            wf[1] = ((float)qr[i][1] - zp[i]) * sc[i];
            wf[2] = ((float)qr[i][2] - zp[i]) * sc[i];
            wf[3] = ((float)qr[i][3] - zp[i]) * sc[i];
            *(v4bf*)(&Ws[buf][row * LDS_PITCH + col]) =
                __builtin_convertvector(wf, v4bf);
        }
    };

    auto compute = [&](int buf) {
#pragma unroll
        for (int kk = 0; kk < BK; kk += 32) {
            v16bf afrag[4];
            v16bf bfrag[2];
#pragma unroll
            for (int mi = 0; mi < 4; ++mi) {
                // A 16x32: lanes<16 K{0..7,16..23}, lanes>=16 K{8..15,24..31}
                const __bf16* base =
                    &As[buf][(warp_m + mi * 16 + l16) * LDS_PITCH + kk + half * 8];
                v8bf lo = *(const v8bf*)(base);
                v8bf hi = *(const v8bf*)(base + 16);
                afrag[mi] = __builtin_shufflevector(lo, hi,
                    0, 1, 2, 3, 4, 5, 6, 7, 8, 9, 10, 11, 12, 13, 14, 15);
            }
#pragma unroll
            for (int ni = 0; ni < 2; ++ni) {
                // B 32x16: lane = column N; lanes<16 K{0..7,16..23}, lanes>=16 K{8..15,24..31}
                const __bf16* base =
                    &Ws[buf][(warp_n + ni * 16 + l16) * LDS_PITCH + kk + half * 8];
                v8bf lo = *(const v8bf*)(base);
                v8bf hi = *(const v8bf*)(base + 16);
                bfrag[ni] = __builtin_shufflevector(lo, hi,
                    0, 1, 2, 3, 4, 5, 6, 7, 8, 9, 10, 11, 12, 13, 14, 15);
            }
#pragma unroll
            for (int mi = 0; mi < 4; ++mi)
#pragma unroll
                for (int ni = 0; ni < 2; ++ni)
                    acc[mi][ni] = __builtin_amdgcn_wmma_f32_16x16x32_bf16(
                        false, afrag[mi], false, bfrag[ni],
                        (short)0, acc[mi][ni], false, false);
        }
    };

    // ---- Software pipeline: load(0); {load(s+1) | compute(s)}; 1 barrier/stage
    stage_load(0);
    stage_store(0);
    __syncthreads();

    int buf = 0;
    for (int kt = 0; kt < K_DIM; kt += BK) {
        const int next = kt + BK;
        const bool have_next = next < K_DIM;

        if (have_next) {
            stage_load(next);                        // global loads in flight
            if (next + BK < K_DIM) {                 // prefetch stage s+2 -> GL2
                const int prow = tid >> 1;
                __builtin_prefetch(x  + (size_t)(m0 + prow) * K_DIM + next + BK, 0, 1);
                __builtin_prefetch(qw + (size_t)(n0 + prow) * K_DIM + next + BK, 0, 1);
            }
        }

        compute(buf);                                // 16 WMMAs from current buf

        if (have_next)
            stage_store(buf ^ 1);                    // convert + fill other buf

        __syncthreads();
        buf ^= 1;
    }

    // ---- Epilogue: bias add + store (C layout: VGPR r -> M=r / r+8, N=lane&15)
#pragma unroll
    for (int ni = 0; ni < 2; ++ni) {
        const int n = n0 + warp_n + ni * 16 + l16;
        const float b = bias[n];
#pragma unroll
        for (int mi = 0; mi < 4; ++mi) {
#pragma unroll
            for (int r = 0; r < 8; ++r) {
                const int m = m0 + warp_m + mi * 16 + half * 8 + r;
                out[(size_t)m * N_DIM + n] = acc[mi][ni][r] + b;
            }
        }
    }
}

extern "C" void kernel_launch(void* const* d_in, const int* in_sizes, int n_in,
                              void* d_out, int out_size, void* d_ws, size_t ws_size,
                              hipStream_t stream) {
    const float* x      = (const float*)d_in[0];
    const int*   qw     = (const int*)d_in[1];
    const float* scales = (const float*)d_in[2];
    const float* zeros  = (const float*)d_in[3];
    const float* bias   = (const float*)d_in[4];
    float*       out    = (float*)d_out;

    dim3 grid(N_DIM / BN, M_DIM / BM);  // (32, 64) = 2048 workgroups
    qlinear_wmma_bf16_kernel<<<grid, dim3(THREADS), 0, stream>>>(
        x, qw, scales, zeros, bias, out);
}
